// TriOrientedMamba_19104014532617
// MI455X (gfx1250) — compile-verified, matrix operand order checked by reading, not run on hardware
//
#include <hip/hip_runtime.h>
#include <hip/hip_bf16.h>
#include <math.h>

// ---------------------------------------------------------------------------
// TriOrientedMamba on MI455X (gfx1250).
// GEMMs via v_wmma_f32_16x16x32_bf16 (bf16 in, f32 accumulate), weight tile
// staged in LDS with GLOBAL_LOAD_ASYNC_TO_LDS_B128 (ASYNCcnt), two-panel
// software pipeline with compile-time buffer selection, reuse-A hints on
// back-to-back identical wmmas.  Selective scan stays fp32 VALU.
// ---------------------------------------------------------------------------

typedef __bf16 bf16_t;
typedef __attribute__((ext_vector_type(16))) __bf16 v16bf;
typedef __attribute__((ext_vector_type(8)))  __bf16 v8bf;
typedef __attribute__((ext_vector_type(8)))  float  v8f;

#define DIMC   128
#define DI_    256
#define DS_    16
#define DTR_   8
#define SEQL   32
#define NTOK   32768   // 32*32*32 tokens
#define NSEQ   1024    // sequences per orientation

#define BROW_STRIDE 40 // padded LDS row stride (elements) for bank spreading

// ---------------------------------------------------------------------------
// WMMA bf16 fragment loader (works for global or LDS generic pointers).
// 16-bit A-matrix 16x32 layout (ISA 7.12.2): lane<16 -> row M=lane holds
// K = 0..7 and 16..23; lane>=16 -> same row, K = 8..15 and 24..31.
// Per lane: two contiguous 16-byte runs at k0+koff and k0+koff+16,
// koff = (lane>>4)*8.  B (=W^T, W row-major [N,K]) mirrors over columns.
// ---------------------------------------------------------------------------
__device__ __forceinline__ v16bf load_frag_bf16(const bf16_t* base, int ld, int k0) {
    const int lane = threadIdx.x & 31;
    const int r    = lane & 15;
    const int koff = (lane >> 4) << 3;
    const bf16_t* p = base + (size_t)r * ld + k0 + koff;
    v8bf lo = *(const v8bf*)(p);
    v8bf hi = *(const v8bf*)(p + 16);
    return __builtin_shufflevector(lo, hi, 0,1,2,3,4,5,6,7,8,9,10,11,12,13,14,15);
}

__device__ __forceinline__ float gelu_exact(float x) {
    return 0.5f * x * (1.0f + erff(x * 0.70710678118654752f));
}

// Async 16-byte global -> LDS copy (gfx1250, tracked with ASYNCcnt).
// LDS byte address = low 32 bits of the flat shared-pointer (flat->LDS
// addressing truncates to addr[31:0], ISA 10.2).
__device__ __forceinline__ void async_copy_b128(const bf16_t* src, bf16_t* lds_dst) {
    unsigned int       ldsa  = (unsigned int)(size_t)lds_dst;
    unsigned long long gaddr = (unsigned long long)(size_t)src;
    asm volatile("global_load_async_to_lds_b128 %0, %1, off"
                 :: "v"(ldsa), "v"(gaddr) : "memory");
}

__device__ __forceinline__ void wait_async_all() {
    asm volatile("s_wait_asynccnt 0x0" ::: "memory");
}

// One 64-wide N panel of wmmas: load all four B fragments from LDS first
// (batched ds_loads -> single dscnt wait), then four back-to-back identical
// wmmas sharing the A operand; reuse-A hint set on the 2nd..4th (legal per
// ISA 7.12: current instruction identical to previous).
__device__ __forceinline__ void wmma_panel(const bf16_t* sBp, v16bf a, v8f acc[4]) {
    v16bf b0 = load_frag_bf16(sBp + 0 * 16 * BROW_STRIDE, BROW_STRIDE, 0);
    v16bf b1 = load_frag_bf16(sBp + 1 * 16 * BROW_STRIDE, BROW_STRIDE, 0);
    v16bf b2 = load_frag_bf16(sBp + 2 * 16 * BROW_STRIDE, BROW_STRIDE, 0);
    v16bf b3 = load_frag_bf16(sBp + 3 * 16 * BROW_STRIDE, BROW_STRIDE, 0);
    acc[0] = __builtin_amdgcn_wmma_f32_16x16x32_bf16(false, a, false, b0, (short)0, acc[0], false, false);
    acc[1] = __builtin_amdgcn_wmma_f32_16x16x32_bf16(false, a, false, b1, (short)0, acc[1], true,  false);
    acc[2] = __builtin_amdgcn_wmma_f32_16x16x32_bf16(false, a, false, b2, (short)0, acc[2], true,  false);
    acc[3] = __builtin_amdgcn_wmma_f32_16x16x32_bf16(false, a, false, b3, (short)0, acc[3], true,  false);
}

// ---------------------------------------------------------------------------
// Generic WMMA GEMM:  C[M,N] = A[M,K] * W[N,K]^T  (+bias, +act, epilogue)
// block = 256 threads = 8 waves; wave w owns rows m0 = (by*8+w)*16 and a
// 16x64 output strip (4 accumulators across N).  The 64xK weight tile is
// shared by all 8 waves -> staged in LDS (async, double-buffered, 32-wide
// K-panels, two panels per loop iteration so buffers are compile-time).
// Requires K % 64 == 0 (holds: 128/256/384).
// OUTMODE 0: f32 row-major (ldc, coloff)
// OUTMODE 1: bf16 row-major (ldc, coloff)
// OUTMODE 2: f32 transposed store + residual:  out[n*M+m] = acc + resid[n*M+m]
// ---------------------------------------------------------------------------
template<int OUTMODE, int ACT, int HAS_BIAS>
__global__ void gemm_wmma(const bf16_t* __restrict__ A, int lda,
                          const bf16_t* __restrict__ W, int ldw,
                          const float*  __restrict__ bias,
                          void* __restrict__ Cout, int ldc, int coloff,
                          const float* __restrict__ resid,
                          int M, int N, int K)
{
    __shared__ bf16_t sB[2][64 * BROW_STRIDE];

    const int tid  = threadIdx.x;
    const int wave = tid >> 5;
    const int lane = tid & 31;
    const int m0 = (blockIdx.y * 8 + wave) * 16;
    const int n0 = blockIdx.x * 64;

    // Per-thread 16B chunk of the 64x32 weight panel.
    const int brow = tid >> 2;          // 0..63
    const int bofs = (tid & 3) * 8;     // element offset within K-panel

    // Prefetch first weight rows (gfx1250 global_prefetch_b8 path).
    __builtin_prefetch(W + (size_t)(n0 + (lane & 15)) * ldw, 0, 3);

    const bf16_t* wsrc  = W + (size_t)(n0 + brow) * ldw + bofs;
    const bf16_t* arow  = A + (size_t)m0 * lda;
    bf16_t* dst0 = &sB[0][brow * BROW_STRIDE + bofs];
    bf16_t* dst1 = &sB[1][brow * BROW_STRIDE + bofs];

    // stage K-panel 0 into buffer 0
    async_copy_b128(wsrc, dst0);
    v16bf a_cur = load_frag_bf16(arow, lda, 0);
    wait_async_all();
    __syncthreads();

    v8f acc[4] = {v8f{}, v8f{}, v8f{}, v8f{}};

    for (int k0 = 0; k0 < K; k0 += 64) {
        // ---- half 1: compute on buf0 (panel k0), prefetch panel k0+32 -> buf1
        async_copy_b128(wsrc + k0 + 32, dst1);
        v16bf a_nxt = load_frag_bf16(arow, lda, k0 + 32);
        wmma_panel(sB[0], a_cur, acc);
        wait_async_all();
        __syncthreads();

        // ---- half 2: compute on buf1 (panel k0+32), prefetch panel k0+64 -> buf0
        const bool more = (k0 + 64 < K);
        v16bf a_nx2;
        if (more) {
            async_copy_b128(wsrc + k0 + 64, dst0);
            a_nx2 = load_frag_bf16(arow, lda, k0 + 64);
        }
        wmma_panel(sB[1], a_nxt, acc);
        if (more) {
            wait_async_all();
            __syncthreads();
            a_cur = a_nx2;
        }
    }

    // C/D layout: lane<16 -> N=lane, M=v;  lane>=16 -> N=lane-16, M=v+8.
    const int ncol = lane & 15;
    const int mrow = (lane >> 4) * 8;
    #pragma unroll
    for (int j = 0; j < 4; ++j) {
        const int n = n0 + j * 16 + ncol;
        float bv = HAS_BIAS ? bias[n] : 0.0f;
        #pragma unroll
        for (int v = 0; v < 8; ++v) {
            const int m = m0 + mrow + v;
            float val = acc[j][v] + bv;
            if (ACT == 1) val = gelu_exact(val);
            if (OUTMODE == 0) {
                ((float*)Cout)[(size_t)m * ldc + coloff + n] = val;
            } else if (OUTMODE == 1) {
                ((bf16_t*)Cout)[(size_t)m * ldc + coloff + n] = (bf16_t)val;
            } else {
                const size_t idx = (size_t)n * M + m;
                ((float*)Cout)[idx] = val + resid[idx];
            }
        }
    }
}

// ---------------------------------------------------------------------------
// Channel LayerNorm:  x (C=128, T=32768) -> xn bf16 (T, 128) token-major.
// One wave per token; lane handles 4 channels.
// ---------------------------------------------------------------------------
__global__ void ln_kernel(const float* __restrict__ x,
                          const float* __restrict__ g,
                          const float* __restrict__ b,
                          bf16_t* __restrict__ xn)
{
    const int wave = threadIdx.x >> 5;
    const int lane = threadIdx.x & 31;
    const int t = blockIdx.x * 8 + wave;

    float v[4];
    float s = 0.f, sq = 0.f;
    #pragma unroll
    for (int i = 0; i < 4; ++i) {
        const int c = lane + 32 * i;
        v[i] = x[(size_t)c * NTOK + t];
        s  += v[i];
        sq += v[i] * v[i];
    }
    #pragma unroll
    for (int off = 16; off > 0; off >>= 1) {
        s  += __shfl_down(s,  off, 32);
        sq += __shfl_down(sq, off, 32);
    }
    s  = __shfl(s,  0, 32);
    sq = __shfl(sq, 0, 32);
    const float mu   = s * (1.0f / DIMC);
    const float var  = sq * (1.0f / DIMC) - mu * mu;
    const float rstd = rsqrtf(var + 1e-5f);
    #pragma unroll
    for (int i = 0; i < 4; ++i) {
        const int c = lane + 32 * i;
        float y = (v[i] - mu) * rstd * g[c] + b[c];
        xn[(size_t)t * DIMC + c] = (bf16_t)y;
    }
}

// ---------------------------------------------------------------------------
// f32 -> bf16 weight conversion
// ---------------------------------------------------------------------------
__global__ void f2bf_kernel(const float* __restrict__ in, bf16_t* __restrict__ out, int n) {
    int i = blockIdx.x * 256 + threadIdx.x;
    if (i < n) out[i] = (bf16_t)in[i];
}

// ---------------------------------------------------------------------------
// Fused causal depthwise conv1d + SiLU + x_proj + dt_proj + selective scan
// + D skip + SiLU(z) gate.  One block (256 threads) per sequence; thread=
// channel, owning the full h[16] state.  xz: (NTOK, 512) f32, cols 0..255 =
// xp, 256..511 = z.  y_g out: (NTOK, 256) bf16 in flat (d,h,w) token order.
// ---------------------------------------------------------------------------
__global__ void mamba_scan_kernel(const float* __restrict__ xz,
                                  const float* __restrict__ conv_w,   // (3,256,1,4)
                                  const float* __restrict__ conv_b,   // (3,256)
                                  const float* __restrict__ x_proj_w, // (3,40,256)
                                  const float* __restrict__ dt_proj_w,// (3,256,8)
                                  const float* __restrict__ dt_proj_b,// (3,256)
                                  const float* __restrict__ A_log,    // (3,256,16)
                                  const float* __restrict__ D_param,  // (3,256)
                                  bf16_t* __restrict__ y_g,
                                  int orient)
{
    const int ch  = threadIdx.x;        // 0..255
    const int seq = blockIdx.x;         // 0..1023
    int base, stride;
    if (orient == 0)      { base = seq;                              stride = 1024; }
    else if (orient == 1) { base = (seq >> 5) * 1024 + (seq & 31);   stride = 32;   }
    else                  { base = seq * 32;                         stride = 1;    }

    const int oc = orient * DI_ + ch;
    float cw[4];
    #pragma unroll
    for (int k = 0; k < 4; ++k) cw[k] = conv_w[(size_t)oc * 4 + k];
    const float cb = conv_b[oc];
    float dtw[DTR_];
    #pragma unroll
    for (int j = 0; j < DTR_; ++j) dtw[j] = dt_proj_w[(size_t)oc * DTR_ + j];
    const float dtb = dt_proj_b[oc];
    float a[DS_];
    #pragma unroll
    for (int s = 0; s < DS_; ++s) a[s] = -__expf(A_log[(size_t)oc * DS_ + s]);
    const float Dp = D_param[oc];
    const float* xpw = x_proj_w + (size_t)orient * 40 * DI_;

    __shared__ float s_xc[DI_];
    __shared__ float s_dbl[48];

    float h[DS_];
    #pragma unroll
    for (int s = 0; s < DS_; ++s) h[s] = 0.f;
    float w0 = 0.f, w1 = 0.f, w2 = 0.f;

    const int wave = ch >> 5, lane = ch & 31;

    for (int pos = 0; pos < SEQL; ++pos) {
        const int token = base + pos * stride;
        const float* row = xz + (size_t)token * 512;

        // causal depthwise conv + SiLU
        const float xp   = row[ch];
        float conv = cb + cw[0] * w0 + cw[1] * w1 + cw[2] * w2 + cw[3] * xp;
        w0 = w1; w1 = w2; w2 = xp;
        const float xc = conv / (1.0f + __expf(-conv));
        s_xc[ch] = xc;
        __syncthreads();

        // dbl[0..39] = xc . x_proj_w[j]  (8 waves x 5 j's, wave-shuffle reduce)
        for (int j = wave; j < 40; j += 8) {
            const float* wr = xpw + (size_t)j * DI_;
            float p = 0.f;
            #pragma unroll
            for (int c2 = 0; c2 < DI_; c2 += 32) p += s_xc[c2 + lane] * wr[c2 + lane];
            #pragma unroll
            for (int off = 16; off > 0; off >>= 1) p += __shfl_down(p, off, 32);
            if (lane == 0) s_dbl[j] = p;
        }
        __syncthreads();

        // dt = softplus(dt_r @ dt_proj_w^T + b)
        float dtr = dtb;
        #pragma unroll
        for (int j = 0; j < DTR_; ++j) dtr += s_dbl[j] * dtw[j];
        const float dt  = (dtr > 20.0f) ? dtr : log1pf(__expf(dtr));
        const float dtx = dt * xc;

        // selective scan step + y = h . C
        float y = 0.f;
        #pragma unroll
        for (int s = 0; s < DS_; ++s) {
            const float dA = __expf(dt * a[s]);
            h[s] = dA * h[s] + dtx * s_dbl[8 + s];
            y += h[s] * s_dbl[24 + s];
        }
        y += Dp * xc;

        // gate with SiLU(z)
        const float z = row[DI_ + ch];
        y *= z / (1.0f + __expf(-z));
        y_g[(size_t)token * DI_ + ch] = (bf16_t)y;
        __syncthreads();   // protect s_xc/s_dbl before next step
    }
}

// ---------------------------------------------------------------------------
// host orchestration
// ---------------------------------------------------------------------------
static inline size_t align256(size_t x) { return (x + 255) & ~(size_t)255; }

extern "C" void kernel_launch(void* const* d_in, const int* in_sizes, int n_in,
                              void* d_out, int out_size, void* d_ws, size_t ws_size,
                              hipStream_t stream)
{
    const float* x          = (const float*)d_in[0];   // (128, 32768)
    const float* norm_g     = (const float*)d_in[1];
    const float* norm_b     = (const float*)d_in[2];
    const float* in_proj_w  = (const float*)d_in[3];   // (3,512,128)
    const float* conv_w     = (const float*)d_in[4];
    const float* conv_b     = (const float*)d_in[5];
    const float* x_proj_w   = (const float*)d_in[6];
    const float* dt_proj_w  = (const float*)d_in[7];
    const float* dt_proj_b  = (const float*)d_in[8];
    const float* A_log      = (const float*)d_in[9];
    const float* D_param    = (const float*)d_in[10];
    const float* out_proj_w = (const float*)d_in[11];  // (3,128,256)
    const float* fusion_w   = (const float*)d_in[12];  // (128,384)
    const float* fusion_b   = (const float*)d_in[13];
    const float* proj_w     = (const float*)d_in[14];  // (128,128)
    const float* proj_b     = (const float*)d_in[15];

    // workspace carve-up
    char* ws = (char*)d_ws;
    size_t off = 0;
    bf16_t* xn     = (bf16_t*)(ws + off); off = align256(off + (size_t)NTOK * DIMC * 2);
    bf16_t* w_in   = (bf16_t*)(ws + off); off = align256(off + (size_t)3 * 512 * 128 * 2);
    bf16_t* w_out  = (bf16_t*)(ws + off); off = align256(off + (size_t)3 * 128 * 256 * 2);
    bf16_t* w_fus  = (bf16_t*)(ws + off); off = align256(off + (size_t)128 * 384 * 2);
    bf16_t* w_proj = (bf16_t*)(ws + off); off = align256(off + (size_t)128 * 128 * 2);
    float*  xz     = (float*) (ws + off); off = align256(off + (size_t)NTOK * 512 * 4);
    bf16_t* y_gb   = (bf16_t*)(ws + off); off = align256(off + (size_t)NTOK * DI_ * 2);
    bf16_t* cat    = (bf16_t*)(ws + off); off = align256(off + (size_t)NTOK * 384 * 2);
    bf16_t* fbuf   = (bf16_t*)(ws + off); off = align256(off + (size_t)NTOK * DIMC * 2);

    // weight f32 -> bf16
    auto cvt = [&](const float* src, bf16_t* dst, int n) {
        f2bf_kernel<<<(n + 255) / 256, 256, 0, stream>>>(src, dst, n);
    };
    cvt(in_proj_w,  w_in,   3 * 512 * 128);
    cvt(out_proj_w, w_out,  3 * 128 * 256);
    cvt(fusion_w,   w_fus,  128 * 384);
    cvt(proj_w,     w_proj, 128 * 128);

    // layernorm -> xn (NTOK, 128) bf16
    ln_kernel<<<NTOK / 8, 256, 0, stream>>>(x, norm_g, norm_b, xn);

    for (int o = 0; o < 3; ++o) {
        // in_proj: xz (NTOK, 512) f32 = xn @ W_o^T
        gemm_wmma<0, 0, 0><<<dim3(512 / 64, NTOK / 128), 256, 0, stream>>>(
            xn, DIMC, w_in + (size_t)o * 512 * 128, DIMC,
            nullptr, xz, 512, 0, nullptr, NTOK, 512, DIMC);

        // conv + silu + x_proj + dt_proj + scan + gate -> y_gb (NTOK, 256) bf16
        mamba_scan_kernel<<<NSEQ, DI_, 0, stream>>>(
            xz, conv_w, conv_b, x_proj_w, dt_proj_w, dt_proj_b,
            A_log, D_param, y_gb, o);

        // out_proj: cat[:, o*128 : (o+1)*128] = y_gb @ W_o^T  (bf16 out, ldc=384)
        gemm_wmma<1, 0, 0><<<dim3(128 / 64, NTOK / 128), 256, 0, stream>>>(
            y_gb, DI_, w_out + (size_t)o * 128 * 256, DI_,
            nullptr, cat, 384, o * 128, nullptr, NTOK, 128, DI_);
    }

    // fusion: f = gelu(cat @ fusion_w^T + b)  (bf16)
    gemm_wmma<1, 1, 1><<<dim3(128 / 64, NTOK / 128), 256, 0, stream>>>(
        cat, 384, w_fus, 384, fusion_b, fbuf, DIMC, 0, nullptr, NTOK, 128, 384);

    // proj + residual, transposed store -> d_out (C, D, H, W) f32
    gemm_wmma<2, 0, 1><<<dim3(128 / 64, NTOK / 128), 256, 0, stream>>>(
        fbuf, DIMC, w_proj, DIMC, proj_b, d_out, 0, 0, x, NTOK, 128, DIMC);

    (void)in_sizes; (void)n_in; (void)out_size; (void)ws_size;
}
